// GPRGNN_52261162058537
// MI455X (gfx1250) — matrix operand order repaired and checked by reading, not compiled, not run on hardware
//
#include <hip/hip_runtime.h>
#include <hip/hip_bf16.h>

typedef __attribute__((ext_vector_type(16))) __bf16 v16bf;
typedef __attribute__((ext_vector_type(8)))  float  v8f;

#define NFEAT  512
#define NHID   256
#define NCLASS 16
#define KPROP  10

// ---------------------------------------------------------------------------
// Pre-pack a row-major f32 weight matrix [K x Ncols] into the per-lane WMMA
// B-operand layout in bf16: for k-step s and column tile nt, lane l
// (l = kh*16 + n) holds the 16 bf16 values B[kb + (i&8)*2 + kh*8 + (i&7)][nt*16 + n],
// stored contiguously (32 bytes per lane -> 2x b128 per fetch).
// ---------------------------------------------------------------------------
__global__ void pack_w_kernel(const float* __restrict__ W, __bf16* __restrict__ Wp,
                              int K, int Ncols)
{
    int idx = blockIdx.x * blockDim.x + threadIdx.x;
    if (idx >= K * Ncols) return;
    int k  = idx / Ncols, c = idx % Ncols;
    int s  = k >> 5, kh = (k >> 3) & 1, ki = (k & 7) | (((k >> 4) & 1) << 3);
    int nt = c >> 4, n  = c & 15;
    int ntiles = Ncols >> 4;
    size_t o = ((((size_t)s * ntiles + nt) * 32) + (kh * 16 + n)) * 16 + ki;
    Wp[o] = (__bf16)W[idx];
}

// ---------------------------------------------------------------------------
// Fused MLP: h = relu(X@W1 + b1) @ W2 + b2, one wave per 16-row tile.
// kb-outer loop, 8 persistent accumulators; all 8 B tiles are fetched in one
// batch (single s_clause) before the 8 back-to-back WMMAs so L2 latency
// overlaps the matrix pipe instead of serializing each WMMA on loadcnt==0.
// GEMM1 result staged in per-wave LDS (16 x 256 bf16) for the GEMM2 A layout.
// ---------------------------------------------------------------------------
__global__ __launch_bounds__(128) void gpr_mlp_kernel(
    const float*  __restrict__ X,
    const __bf16* __restrict__ W1p, const float* __restrict__ b1,
    const __bf16* __restrict__ W2p, const float* __restrict__ b2,
    const float*  __restrict__ gamma,
    float* __restrict__ hcur, float* __restrict__ out, int nrows)
{
    __shared__ __align__(16) __bf16 lds[4][16 * NHID];   // 4 waves * 8KB

    const int wave = threadIdx.x >> 5;
    const int lane = threadIdx.x & 31;
    const int tile = blockIdx.x * 4 + wave;
    if (tile * 16 >= nrows) return;            // wave-uniform guard (EXEC all-1s)

    const int rm = tile * 16;
    const int m  = lane & 15;                  // A row / B,C column
    const int kh = lane >> 4;                  // K-half selector
    const int n  = lane & 15;
    const int mh = lane >> 4;
    __bf16* my = lds[wave];

    // ---- GEMM1: two passes of 8 column tiles each ----
    for (int half = 0; half < 2; ++half) {
        v8f acc[8];
        #pragma unroll
        for (int t = 0; t < 8; ++t) acc[t] = (v8f){};

        for (int s = 0; s < NFEAT / 32; ++s) {
            const int kb = s * 32;

            // A (16x32 bf16): lane m = row; kh selects K {0..7,16..23} / {8..15,24..31}
            const float* xr = X + (size_t)(rm + m) * NFEAT + kb + kh * 8;
            float4 x0 = *(const float4*)(xr);
            float4 x1 = *(const float4*)(xr + 4);
            float4 x2 = *(const float4*)(xr + 16);
            float4 x3 = *(const float4*)(xr + 20);

            // batch all 8 packed-B fetches first (one clause, staggered waits)
            v16bf bb[8];
            #pragma unroll
            for (int t = 0; t < 8; ++t) {
                const int nt = half * 8 + t;
                bb[t] = *(const v16bf*)(W1p +
                         ((((size_t)s * (NHID / 16) + nt) * 32) + lane) * 16);
            }

            v16bf a;
            a[0]  = (__bf16)x0.x; a[1]  = (__bf16)x0.y; a[2]  = (__bf16)x0.z; a[3]  = (__bf16)x0.w;
            a[4]  = (__bf16)x1.x; a[5]  = (__bf16)x1.y; a[6]  = (__bf16)x1.z; a[7]  = (__bf16)x1.w;
            a[8]  = (__bf16)x2.x; a[9]  = (__bf16)x2.y; a[10] = (__bf16)x2.z; a[11] = (__bf16)x2.w;
            a[12] = (__bf16)x3.x; a[13] = (__bf16)x3.y; a[14] = (__bf16)x3.z; a[15] = (__bf16)x3.w;

            #pragma unroll
            for (int t = 0; t < 8; ++t) {
                acc[t] = __builtin_amdgcn_wmma_f32_16x16x32_bf16(
                    false, a, false, bb[t], (short)0, acc[t], false, false);
            }
        }

        // bias + relu -> LDS (C layout: lane = col n, rows mh*8 + r)
        #pragma unroll
        for (int t = 0; t < 8; ++t) {
            const int   nt   = half * 8 + t;
            const float bias = b1[nt * 16 + n];
            #pragma unroll
            for (int r = 0; r < 8; ++r) {
                float v = acc[t][r] + bias;
                v = v > 0.0f ? v : 0.0f;
                my[(mh * 8 + r) * NHID + nt * 16 + n] = (__bf16)v;
            }
        }
    }
    // per-wave LDS slice; DS ops are in-order within a wave -> no barrier

    // ---- GEMM2: h[16x16] = H1 @ W2 + b2 ----
    v8f acc2 = {};
    #pragma unroll
    for (int s = 0; s < NHID / 32; ++s) {
        const int kb = s * 32;
        union { v16bf v; uint4 q[2]; } au;
        au.q[0] = *(const uint4*)(my + m * NHID + kb + kh * 8);        // ds_load_b128
        au.q[1] = *(const uint4*)(my + m * NHID + kb + 16 + kh * 8);   // ds_load_b128
        v16bf b = *(const v16bf*)(W2p + ((size_t)s * 32 + lane) * 16);
        acc2 = __builtin_amdgcn_wmma_f32_16x16x32_bf16(
            false, au.v, false, b, (short)0, acc2, false, false);
    }

    const float g0    = gamma[0];
    const float bias2 = b2[n];
    #pragma unroll
    for (int r = 0; r < 8; ++r) {
        float  v   = acc2[r] + bias2;
        size_t idx = (size_t)(rm + mh * 8 + r) * NCLASS + n;
        hcur[idx] = v;
        out[idx]  = g0 * v;
    }
}

// ---------------------------------------------------------------------------
// Graph-propagation helper kernels
// ---------------------------------------------------------------------------
__global__ void zero_kernel(float* __restrict__ p, int n)
{
    int i = blockIdx.x * blockDim.x + threadIdx.x;
    if (i < n) p[i] = 0.0f;
}

__global__ void deg_kernel(const int* __restrict__ row, float* __restrict__ deg, int ne)
{
    int e = blockIdx.x * blockDim.x + threadIdx.x;
    if (e < ne) atomicAdd(&deg[row[e]], 1.0f);
}

__global__ void dis_kernel(float* __restrict__ deg, int n)
{
    int i = blockIdx.x * blockDim.x + threadIdx.x;
    if (i < n) {
        float d = deg[i];
        deg[i] = d > 0.0f ? rsqrtf(d) : 0.0f;   // deg -> deg^{-1/2} in place
    }
}

__global__ void norm_kernel(const int* __restrict__ row, const int* __restrict__ col,
                            const float* __restrict__ dis, float* __restrict__ norm, int ne)
{
    int e = blockIdx.x * blockDim.x + threadIdx.x;
    if (e < ne) norm[e] = dis[row[e]] * dis[col[e]];
}

// 16 lanes per edge: contiguous 64B gather of h_in[row] and contiguous 64B
// atomic scatter into h_out[col] (coalescable by the L2 atomic units).
__global__ void prop_kernel(const int* __restrict__ row, const int* __restrict__ col,
                            const float* __restrict__ norm,
                            const float* __restrict__ hin, float* __restrict__ hout,
                            int ne)
{
    int tid = blockIdx.x * blockDim.x + threadIdx.x;
    int e   = tid >> 4;
    if (e >= ne) return;
    int f = tid & 15;
    int r = row[e];
    int c = col[e];
    float v = norm[e] * hin[(size_t)r * NCLASS + f];
    atomicAdd(&hout[(size_t)c * NCLASS + f], v);
}

__global__ void axpy_kernel(const float* __restrict__ hin, float* __restrict__ out,
                            const float* __restrict__ gamma, int k, int n)
{
    int i = blockIdx.x * blockDim.x + threadIdx.x;
    if (i < n) out[i] += gamma[k] * hin[i];
}

// ---------------------------------------------------------------------------
static inline size_t alignup(size_t x) { return (x + 255) & ~(size_t)255; }

extern "C" void kernel_launch(void* const* d_in, const int* in_sizes, int n_in,
                              void* d_out, int out_size, void* d_ws, size_t ws_size,
                              hipStream_t stream)
{
    const float* X     = (const float*)d_in[0];
    const int*   ei    = (const int*)  d_in[1];
    const float* W1    = (const float*)d_in[2];
    const float* b1    = (const float*)d_in[3];
    const float* W2    = (const float*)d_in[4];
    const float* b2    = (const float*)d_in[5];
    const float* gamma = (const float*)d_in[6];
    float*       out   = (float*)d_out;

    const int N = in_sizes[0] / NFEAT;      // 100000
    const int E = in_sizes[1] / 2;          // 3200000
    const int* row = ei;
    const int* col = ei + E;

    // workspace carve-out
    char*   ws      = (char*)d_ws;
    size_t  offDeg  = 0;
    size_t  offNorm = alignup(offDeg  + (size_t)N * 4);
    size_t  offA    = alignup(offNorm + (size_t)E * 4);
    size_t  offB    = alignup(offA    + (size_t)N * NCLASS * 4);
    size_t  offW1p  = alignup(offB    + (size_t)N * NCLASS * 4);
    size_t  offW2p  = alignup(offW1p  + (size_t)NFEAT * NHID * 2);
    float*  deg  = (float*) (ws + offDeg);
    float*  norm = (float*) (ws + offNorm);
    float*  bufA = (float*) (ws + offA);
    float*  bufB = (float*) (ws + offB);
    __bf16* W1p  = (__bf16*)(ws + offW1p);
    __bf16* W2p  = (__bf16*)(ws + offW2p);

    const int nf  = N * NCLASS;
    const int tB  = 256;
    const int gN  = (N  + tB - 1) / tB;
    const int gE  = (E  + tB - 1) / tB;
    const int gNF = (nf + tB - 1) / tB;
    const int gEF = (E * NCLASS + tB - 1) / tB;

    // 0) pack weights into WMMA B-operand bf16 layout (one-time, L2 resident)
    pack_w_kernel<<<(NFEAT * NHID + tB - 1) / tB, tB, 0, stream>>>(W1, W1p, NFEAT, NHID);
    pack_w_kernel<<<(NHID * NCLASS + tB - 1) / tB, tB, 0, stream>>>(W2, W2p, NHID, NCLASS);

    // 1) fused MLP -> bufA = h, out = gamma[0]*h
    {
        int tiles  = (N + 15) / 16;
        int blocks = (tiles + 3) / 4;
        gpr_mlp_kernel<<<blocks, 128, 0, stream>>>(X, W1p, b1, W2p, b2, gamma,
                                                   bufA, out, N);
    }

    // 2) normalization coefficients
    zero_kernel<<<gN, tB, 0, stream>>>(deg, N);
    deg_kernel <<<gE, tB, 0, stream>>>(row, deg, E);
    dis_kernel <<<gN, tB, 0, stream>>>(deg, N);
    norm_kernel<<<gE, tB, 0, stream>>>(row, col, deg, norm, E);

    // 3) K propagation steps with ping-pong buffers
    float* cur = bufA;
    float* nxt = bufB;
    for (int k = 1; k <= KPROP; ++k) {
        zero_kernel<<<gNF, tB, 0, stream>>>(nxt, nf);
        prop_kernel<<<gEF, tB, 0, stream>>>(row, col, norm, cur, nxt, E);
        axpy_kernel<<<gNF, tB, 0, stream>>>(nxt, out, gamma, k, nf);
        float* t = cur; cur = nxt; nxt = t;
    }
    (void)out_size; (void)ws_size; (void)n_in;
}
